// FB_87746181858020
// MI455X (gfx1250) — compile-verified, matrix operand order checked by reading, not run on hardware
//
#include <hip/hip_runtime.h>
#include <hip/hip_bf16.h>

typedef __attribute__((ext_vector_type(16))) _Float16 v16h;
typedef __attribute__((ext_vector_type(8)))  float    v8f;
typedef __attribute__((ext_vector_type(4)))  float    f4;

#define D_IN      592
#define NB        8       // filter banks
#define TPB       256     // 8 waves (wave32)
#define SPB       16      // samples per block (WMMA M)
#define XA_STRIDE 600     // padded LDS row stride (floats), 16B-aligned rows
#define P2_STRIDE 40
#define A1_STRIDE 24
#define DQ        (D_IN / 4)   // 148 float4s per row

// Element->K mapping for 16-bit A/B operands of V_WMMA_*_16X16X32 (wave32):
// lanes 0-15 : elems 0-7 -> K 0-7,  elems 8-15 -> K 16-23
// lanes 16-31: elems 0-7 -> K 8-15, elems 8-15 -> K 24-31
__device__ __forceinline__ int kmap(int e, int half) {
  return ((e < 8) ? 0 : 16) + half * 8 + (e & 7);
}

__global__ __launch_bounds__(TPB)
void fbank_fused_kernel(const float* __restrict__ x,
                        const float* __restrict__ c1w, const float* __restrict__ c1b,
                        const float* __restrict__ c2w, const float* __restrict__ c2b,
                        const float* __restrict__ l1w, const float* __restrict__ l1b,
                        const float* __restrict__ l2w, const float* __restrict__ l2b,
                        const float* __restrict__ l3w, const float* __restrict__ l3b,
                        float* __restrict__ out, int B)
{
  __shared__ float lds_xa[SPB * XA_STRIDE];  // residual x_aux, updated in place per bank
  __shared__ float lds_p2[SPB * P2_STRIDE];  // pooled conv output (16 x 36)
  __shared__ float lds_a1[SPB * A1_STRIDE];  // lin1 activations (16 x 20)
  __shared__ float lds_a2[SPB * 16];         // lin3 partial products
  __shared__ float lds_f0[SPB];

  const int tid = threadIdx.x;
  const int s0  = blockIdx.x * SPB;
  const size_t plane  = (size_t)B * D_IN;
  const size_t xfBase = (size_t)NB * plane;
  const size_t xsBase = 2 * xfBase;
  const size_t f0Base = 3 * xfBase;

  // ---- Stage 0: load x tile into LDS (vectorized b128) ----
  for (int t = tid; t < SPB * DQ; t += TPB) {
    int s = t / DQ, q = t - s * DQ;
    size_t S = (size_t)(s0 + s);
    f4 v = {0.f, 0.f, 0.f, 0.f};
    if (S < (size_t)B) v = *(const f4*)(x + S * (size_t)D_IN + 4 * q);
    *(f4*)&lds_xa[s * XA_STRIDE + 4 * q] = v;
  }
  __syncthreads();

  const float inv2w2 = 0.02f;  // 1 / (2 * 5^2)

  for (int i = 0; i < NB; ++i) {
    // ---- conv weights (uniform across block -> scalar loads) ----
    float w1[3][3], b1[3], w2[3][3];
    for (int oc = 0; oc < 3; ++oc) {
      b1[oc] = c1b[i * 3 + oc];
      for (int k = 0; k < 3; ++k) w1[oc][k] = c1w[i * 9 + oc * 3 + k];
    }
    float b2 = c2b[i];
    for (int ic = 0; ic < 3; ++ic)
      for (int k = 0; k < 3; ++k) w2[ic][k] = c2w[i * 9 + ic * 3 + k];

    // ---- Stage 1: fused conv1+pool+relu -> conv2+pool+relu  (16x36 outputs) ----
    // pooled2[s][j] depends only on lds_xa[s][16j .. 16j+30]; no boundary cases.
    for (int t = tid; t < SPB * 36; t += TPB) {
      int s = t / 36, j = t - s * 36;
      const float* xr = &lds_xa[s * XA_STRIDE + 16 * j];
      float p1[3][7];
      for (int q = 0; q < 7; ++q)
        for (int oc = 0; oc < 3; ++oc) {
          float m = -3.0e38f;
          for (int tt = 0; tt < 3; ++tt) {
            int bse = 4 * q + 2 * tt;
            float c = __builtin_fmaf(w1[oc][0], xr[bse], b1[oc]);
            c = __builtin_fmaf(w1[oc][1], xr[bse + 1], c);
            c = __builtin_fmaf(w1[oc][2], xr[bse + 2], c);
            m = fmaxf(m, c);
          }
          p1[oc][q] = fmaxf(m, 0.f);
        }
      float m2 = -3.0e38f;
      for (int tt = 0; tt < 3; ++tt) {
        float acc = b2;
        for (int ic = 0; ic < 3; ++ic)
          for (int k = 0; k < 3; ++k)
            acc = __builtin_fmaf(p1[ic][2 * tt + k], w2[ic][k], acc);
        m2 = fmaxf(m2, acc);
      }
      lds_p2[s * P2_STRIDE + j] = fmaxf(m2, 0.f);
    }
    __syncthreads();

    // ---- Stage 2: lin1 (36->20) via WMMA f32_16x16x32_f16, wave 0 only ----
    {
      const int half = tid >> 4, col = tid & 15;
      if (tid < 32) {
        v16h a0, a1, b00, b10, b01, b11;
        const float* W1 = l1w + (size_t)i * 720;   // [20][36]
        for (int e = 0; e < 16; ++e) {
          int kk = kmap(e, half);                  // 0..31
          a0[e]  = (_Float16)lds_p2[col * P2_STRIDE + kk];
          a1[e]  = (_Float16)((kk < 4) ? lds_p2[col * P2_STRIDE + 32 + kk] : 0.f);
          b00[e] = (_Float16)W1[col * 36 + kk];
          b10[e] = (_Float16)((kk < 4) ? W1[col * 36 + 32 + kk] : 0.f);
          int N1 = 16 + col;
          b01[e] = (_Float16)((N1 < 20) ? W1[N1 * 36 + kk] : 0.f);
          b11[e] = (_Float16)((N1 < 20 && kk < 4) ? W1[N1 * 36 + 32 + kk] : 0.f);
        }
        v8f c0 = {}, c1 = {};
        c0 = __builtin_amdgcn_wmma_f32_16x16x32_f16(false, a0, false, b00, (short)0, c0, false, false);
        c0 = __builtin_amdgcn_wmma_f32_16x16x32_f16(false, a1, false, b10, (short)0, c0, false, false);
        c1 = __builtin_amdgcn_wmma_f32_16x16x32_f16(false, a0, false, b01, (short)0, c1, false, false);
        c1 = __builtin_amdgcn_wmma_f32_16x16x32_f16(false, a1, false, b11, (short)0, c1, false, false);
        float bias0 = l1b[i * 20 + col];
        float bias1 = (16 + col < 20) ? l1b[i * 20 + 16 + col] : 0.f;
        for (int v = 0; v < 8; ++v) {            // D layout: row M = v + 8*half, col N = lane%16
          int m = v + 8 * half;
          lds_a1[m * A1_STRIDE + col] = fmaxf(c0[v] + bias0, 0.f);
          if (16 + col < 20)
            lds_a1[m * A1_STRIDE + 16 + col] = fmaxf(c1[v] + bias1, 0.f);
        }
      }
      __syncthreads();

      // ---- Stage 3: lin2 (20->10) via WMMA + lin3 partials ----
      if (tid < 32) {
        v16h a2, bb2;
        const float* W2 = l2w + (size_t)i * 200;   // [10][20]
        for (int e = 0; e < 16; ++e) {
          int kk = kmap(e, half);
          a2[e]  = (_Float16)((kk < 20) ? lds_a1[col * A1_STRIDE + kk] : 0.f);
          bb2[e] = (_Float16)((kk < 20 && col < 10) ? W2[col * 20 + kk] : 0.f);
        }
        v8f c2 = {};
        c2 = __builtin_amdgcn_wmma_f32_16x16x32_f16(false, a2, false, bb2, (short)0, c2, false, false);
        float bias2 = (col < 10) ? l2b[i * 10 + col] : 0.f;
        float w3    = (col < 10) ? l3w[i * 10 + col] : 0.f;
        for (int v = 0; v < 8; ++v) {
          float y2 = (col < 10) ? fmaxf(c2[v] + bias2, 0.f) : 0.f;
          lds_a2[(v + 8 * half) * 16 + col] = y2 * w3;
        }
      }
      __syncthreads();

      // ---- Stage 4: lin3 reduce + sigmoid -> f0 ----
      if (tid < 16) {
        float acc = l3b[i];
        for (int n = 0; n < 16; ++n) acc += lds_a2[tid * 16 + n];
        float f0v = (float)D_IN / (1.f + __expf(-acc));   // D * sigmoid(z)
        lds_f0[tid] = f0v;
        size_t S = (size_t)(s0 + tid);
        if (S < (size_t)B) out[f0Base + (size_t)i * B + S] = f0v;
      }
      __syncthreads();
    }

    // ---- Stage 5: Gaussian H, xf, xs; stream out NT b128; update residual ----
    for (int t = tid; t < SPB * DQ; t += TPB) {
      int s = t / DQ, q = t - s * DQ;
      int d = 4 * q;
      f4 xa = *(const f4*)&lds_xa[s * XA_STRIDE + d];
      float f0v = lds_f0[s];
      f4 H, xf;
      for (int u = 0; u < 4; ++u) {
        float diff = (float)(d + u) - f0v;
        H[u]  = __expf(-diff * diff * inv2w2);
        xf[u] = xa[u] * H[u];
      }
      size_t S = (size_t)(s0 + s);
      if (S < (size_t)B) {
        size_t rb = ((size_t)i * B + S) * D_IN + d;
        __builtin_nontemporal_store(H,  (f4*)(out + rb));
        __builtin_nontemporal_store(xf, (f4*)(out + xfBase + rb));
        __builtin_nontemporal_store(xa, (f4*)(out + xsBase + rb));
      }
      *(f4*)&lds_xa[s * XA_STRIDE + d] = xa - xf;   // x_aux_{i+1} = x_aux_i - xf_i
    }
    __syncthreads();
  }
}

extern "C" void kernel_launch(void* const* d_in, const int* in_sizes, int n_in,
                              void* d_out, int out_size, void* d_ws, size_t ws_size,
                              hipStream_t stream) {
  const float* x   = (const float*)d_in[0];
  const float* c1w = (const float*)d_in[1];
  const float* c1b = (const float*)d_in[2];
  const float* c2w = (const float*)d_in[3];
  const float* c2b = (const float*)d_in[4];
  const float* l1w = (const float*)d_in[5];
  const float* l1b = (const float*)d_in[6];
  const float* l2w = (const float*)d_in[7];
  const float* l2b = (const float*)d_in[8];
  const float* l3w = (const float*)d_in[9];
  const float* l3b = (const float*)d_in[10];
  int B = in_sizes[0] / D_IN;
  int grid = (B + SPB - 1) / SPB;
  fbank_fused_kernel<<<grid, TPB, 0, stream>>>(x, c1w, c1b, c2w, c2b,
                                               l1w, l1b, l2w, l2b, l3w, l3b,
                                               (float*)d_out, B);
}
